// Mamba_10926396801232
// MI455X (gfx1250) — compile-verified, matrix operand order checked by reading, not run on hardware
//
#include <hip/hip_runtime.h>
#include <hip/hip_bf16.h>
#include <math.h>

#define D_MODEL 768
#define D_INNER 1536
#define N_LAYERS 4
#define VOCAB   32000
#define D_STATE 16
#define DT_RANK 48
#define DT_PAD  64
#define D_CONV  4
#define SEQ_L   1024

typedef __attribute__((ext_vector_type(16))) __bf16 v16bf;
typedef __attribute__((ext_vector_type(8)))  float  v8f;

// ---------------------------------------------------------------------------
// Embedding gather: x[t,:] = emb[ids[t],:]
// ---------------------------------------------------------------------------
__global__ void embed_kernel(const int* __restrict__ ids,
                             const float* __restrict__ emb,
                             float* __restrict__ x) {
    int t = blockIdx.x;
    int row = ids[t];
    for (int i = threadIdx.x; i < D_MODEL; i += blockDim.x)
        x[(size_t)t * D_MODEL + i] = emb[(size_t)row * D_MODEL + i];
}

// ---------------------------------------------------------------------------
// RMSNorm over last dim (768), one block per token
// ---------------------------------------------------------------------------
__global__ void rmsnorm_kernel(const float* __restrict__ x,
                               const float* __restrict__ w,
                               float* __restrict__ out) {
    __shared__ float red[8];
    int t = blockIdx.x;
    const float* xi = x + (size_t)t * D_MODEL;
    float s = 0.f;
    for (int i = threadIdx.x; i < D_MODEL; i += blockDim.x) { float v = xi[i]; s += v * v; }
    #pragma unroll
    for (int off = 16; off > 0; off >>= 1) s += __shfl_down(s, off, 32);
    int wave = threadIdx.x >> 5;
    if ((threadIdx.x & 31) == 0) red[wave] = s;
    __syncthreads();
    if (threadIdx.x == 0) {
        float tot = 0.f;
        int nw = (int)(blockDim.x >> 5);
        for (int i = 0; i < nw; ++i) tot += red[i];
        red[0] = rsqrtf(tot / (float)D_MODEL + 1e-15f);
    }
    __syncthreads();
    float scale = red[0];
    for (int i = threadIdx.x; i < D_MODEL; i += blockDim.x)
        out[(size_t)t * D_MODEL + i] = xi[i] * scale * w[i];
}

// ---------------------------------------------------------------------------
// GEMM: C[M,N] = A[M,K] @ B (+optional residual R), bf16x2 split on
// v_wmma_f32_16x16x32_bf16.
//   BT=false: B is (K,N) row-major;  BT=true: B is (N,K) row-major.
//   Block = 8 waves = 8 consecutive M-tiles sharing NTILE N-tiles.
//   Each wave computes a 16 x (NTILE*16) strip: A fragment converted once
//   per k-step, reused for 3*NTILE WMMAs.  B staged via LDS, coalesced,
//   bank-conflict-padded.  K % 32 == 0, M % 128 == 0, N % (NTILE*16) == 0.
//   No divergent branches: EXEC all-1s at every WMMA.
// ---------------------------------------------------------------------------
template <bool BT, int NTILE>
__global__ void gemm_bf16x2(const float* __restrict__ A, int lda,
                            const float* __restrict__ B, int ldb,
                            float* __restrict__ C, int ldc,
                            const float* __restrict__ R, int ldr,
                            int nblocks_n, int K) {
    constexpr int NCOLS = NTILE * 16;
    __shared__ float bT[32][NCOLS + 1];          // +1: bank-conflict padding

    int bn     = (int)blockIdx.x % nblocks_n;
    int tmBase = ((int)blockIdx.x / nblocks_n) << 3;   // 8 m-tiles per block
    int wave = (int)threadIdx.x >> 5;
    int lane = (int)threadIdx.x & 31;
    int m0  = (tmBase + wave) << 4;
    int n0b = bn * NCOLS;
    int r = lane & 15, half = lane >> 4;
    const float* Arow = A + (size_t)(m0 + r) * lda;

    int base0 = half * 8;        // K sub-block for vector elems 0..7
    int base1 = 16 + half * 8;   // K sub-block for vector elems 8..15

    // this thread's q=0 coop-load coordinates (used for B prefetch)
    int pf_kk, pf_c;
    if (BT) { pf_c = (int)threadIdx.x >> 5;      pf_kk = (int)threadIdx.x & 31; }
    else    { pf_c = (int)threadIdx.x % NCOLS;   pf_kk = (int)threadIdx.x / NCOLS; }

    v8f acc[NTILE];
    #pragma unroll
    for (int nt = 0; nt < NTILE; ++nt) acc[nt] = (v8f){};

    for (int k0 = 0; k0 < K; k0 += 32) {
        __syncthreads();
        // cooperative B-tile load: bT[kk][c] = B[k0+kk, n0b+c]
        #pragma unroll
        for (int q = 0; q < (32 * NCOLS) / 256; ++q) {
            int i = (int)threadIdx.x + q * 256;
            int kk, c;
            if (BT) { c = i >> 5;       kk = i & 31; }      // contiguous in K
            else    { c = i % NCOLS;    kk = i / NCOLS; }   // contiguous in N
            bT[kk][c] = BT ? B[(size_t)(n0b + c) * ldb + k0 + kk]
                           : B[(size_t)(k0 + kk) * ldb + n0b + c];
        }
        __syncthreads();

        // prefetch next k-step's A fragment and B coop-load slot
        // (global_prefetch_b8: whole cacheline, no counter, no VGPR return)
        if (k0 + 32 < K) {
            __builtin_prefetch(Arow + k0 + 32 + base0, 0, 3);
            __builtin_prefetch(BT ? (B + (size_t)(n0b + pf_c) * ldb + k0 + 32 + pf_kk)
                                  : (B + (size_t)(k0 + 32 + pf_kk) * ldb + n0b + pf_c),
                               0, 3);
        }

        // A fragment: two 8-float contiguous runs -> four b128 loads
        float4 a0 = *(const float4*)(Arow + k0 + base0);
        float4 a1 = *(const float4*)(Arow + k0 + base0 + 4);
        float4 a2 = *(const float4*)(Arow + k0 + base1);
        float4 a3 = *(const float4*)(Arow + k0 + base1 + 4);
        float av[16] = { a0.x, a0.y, a0.z, a0.w,  a1.x, a1.y, a1.z, a1.w,
                         a2.x, a2.y, a2.z, a2.w,  a3.x, a3.y, a3.z, a3.w };
        v16bf ah, al;
        #pragma unroll
        for (int j = 0; j < 16; ++j) {
            __bf16 h = (__bf16)av[j];
            ah[j] = h;
            al[j] = (__bf16)(av[j] - (float)h);
        }

        #pragma unroll
        for (int nt = 0; nt < NTILE; ++nt) {
            int cc = nt * 16 + r;
            float bv[16];
            #pragma unroll
            for (int j = 0; j < 8; ++j) {
                bv[j]     = bT[base0 + j][cc];
                bv[j + 8] = bT[base1 + j][cc];
            }
            v16bf bh, bl;
            #pragma unroll
            for (int j = 0; j < 16; ++j) {
                __bf16 g = (__bf16)bv[j];
                bh[j] = g;
                bl[j] = (__bf16)(bv[j] - (float)g);
            }
            // A*B ~= ah*bh + ah*bl + al*bh  (al*bl below f32 precision)
            acc[nt] = __builtin_amdgcn_wmma_f32_16x16x32_bf16(false, ah, false, bh, (short)0, acc[nt], false, false);
            acc[nt] = __builtin_amdgcn_wmma_f32_16x16x32_bf16(false, ah, false, bl, (short)0, acc[nt], false, false);
            acc[nt] = __builtin_amdgcn_wmma_f32_16x16x32_bf16(false, al, false, bh, (short)0, acc[nt], false, false);
        }
    }

    // C/D layout: VGPR p, lanes 0-15 -> M=p, lanes 16-31 -> M=p+8
    #pragma unroll
    for (int nt = 0; nt < NTILE; ++nt) {
        #pragma unroll
        for (int p = 0; p < 8; ++p) {
            int row = m0 + p + half * 8;
            int col = n0b + nt * 16 + r;
            float v = acc[nt][p];
            if (R) v += R[(size_t)row * ldr + col];
            C[(size_t)row * ldc + col] = v;
        }
    }
}

// ---------------------------------------------------------------------------
// Causal depthwise conv (D_CONV=4) + bias + SiLU.  u = xr[:, :D_INNER]
// ---------------------------------------------------------------------------
__global__ void conv_silu_kernel(const float* __restrict__ xr,
                                 const float* __restrict__ cw,
                                 const float* __restrict__ cb,
                                 float* __restrict__ uc) {
    int idx = blockIdx.x * blockDim.x + threadIdx.x;
    if (idx >= SEQ_L * D_INNER) return;
    int t = idx / D_INNER, d = idx - t * D_INNER;
    float acc = cb[d];
    #pragma unroll
    for (int k = 0; k < D_CONV; ++k) {
        int tt = t + k - (D_CONV - 1);
        if (tt >= 0) acc += xr[(size_t)tt * (2 * D_INNER) + d] * cw[k * D_INNER + d];
    }
    uc[idx] = acc / (1.f + __expf(-acc));   // silu
}

// ---------------------------------------------------------------------------
// dtp[t, 0:64] = { mid[t, 0:48], zeros } (A side of dt_proj, guard-free K=64)
// ---------------------------------------------------------------------------
__global__ void pad_dt_kernel(const float* __restrict__ mid,
                              float* __restrict__ dtp) {
    int i = blockIdx.x * blockDim.x + threadIdx.x;
    if (i >= SEQ_L * DT_PAD) return;
    int t = i >> 6, c = i & 63;
    dtp[i] = (c < DT_RANK) ? mid[(size_t)t * (DT_RANK + 2 * D_STATE) + c] : 0.f;
}

// ---------------------------------------------------------------------------
// Zero-padded dt_proj weights: wp[layer, 0:64, :] = { w[layer, 0:48, :], 0 }.
// Run once per launch; removes all OOB reads from the dt_proj GEMM.
// ---------------------------------------------------------------------------
__global__ void pad_dtw_kernel(const float* __restrict__ w,
                               float* __restrict__ wp) {
    int i = blockIdx.x * blockDim.x + threadIdx.x;
    if (i >= N_LAYERS * DT_PAD * D_INNER) return;
    int layer = i / (DT_PAD * D_INNER);
    int rem   = i - layer * (DT_PAD * D_INNER);
    int k = rem / D_INNER, c = rem - k * D_INNER;
    wp[i] = (k < DT_RANK)
          ? w[(size_t)layer * DT_RANK * D_INNER + (size_t)k * D_INNER + c]
          : 0.f;
}

// ---------------------------------------------------------------------------
// delta = softplus(delta_pre + bdt)
// ---------------------------------------------------------------------------
__global__ void softplus_kernel(float* __restrict__ delta,
                                const float* __restrict__ bdt) {
    int idx = blockIdx.x * blockDim.x + threadIdx.x;
    if (idx >= SEQ_L * D_INNER) return;
    int d = idx % D_INNER;
    float v = delta[idx] + bdt[d];
    delta[idx] = (v > 20.f) ? v : log1pf(__expf(v));
}

// ---------------------------------------------------------------------------
// Selective scan: one thread per channel d, 16-state recurrence in registers,
// fused with  + u*D  and  * silu(res)  gating.
// ---------------------------------------------------------------------------
__global__ void scan_kernel(const float* __restrict__ delta,
                            const float* __restrict__ uc,
                            const float* __restrict__ mid,
                            const float* __restrict__ xr,
                            const float* __restrict__ A_log,
                            const float* __restrict__ Dp,
                            float* __restrict__ y) {
    int d = blockIdx.x * blockDim.x + threadIdx.x;
    if (d >= D_INNER) return;
    float a[D_STATE], h[D_STATE];
    #pragma unroll
    for (int n = 0; n < D_STATE; ++n) {
        a[n] = -__expf(A_log[(size_t)d * D_STATE + n]);  // A = -exp(A_log)
        h[n] = 0.f;
    }
    float Dd = Dp[d];
    for (int t = 0; t < SEQ_L; ++t) {
        float dt = delta[(size_t)t * D_INNER + d];
        float ut = uc[(size_t)t * D_INNER + d];
        float dtu = dt * ut;
        const float* m = mid + (size_t)t * (DT_RANK + 2 * D_STATE);
        float acc = 0.f;
        #pragma unroll
        for (int n = 0; n < D_STATE; ++n) {
            float dA = __expf(dt * a[n]);
            h[n] = dA * h[n] + dtu * m[DT_RANK + n];          // B_t[n]
            acc += h[n] * m[DT_RANK + D_STATE + n];           // C_t[n]
        }
        float yy = acc + ut * Dd;
        float res = xr[(size_t)t * (2 * D_INNER) + D_INNER + d];
        yy *= res / (1.f + __expf(-res));                     // * silu(res)
        y[(size_t)t * D_INNER + d] = yy;
    }
}

// ---------------------------------------------------------------------------
// Host side
// ---------------------------------------------------------------------------
template <bool BT, int NTILE>
static void launch_gemm(const float* A, int lda, const float* B, int ldb,
                        float* C, int ldc, const float* R, int ldr,
                        int M, int N, int K, hipStream_t s) {
    int nblocks_n = N / (NTILE * 16);
    int blocks    = nblocks_n * (M / 128);   // 8 m-tiles (8 waves) per block
    gemm_bf16x2<BT, NTILE><<<blocks, 256, 0, s>>>(A, lda, B, ldb, C, ldc, R, ldr,
                                                  nblocks_n, K);
}

extern "C" void kernel_launch(void* const* d_in, const int* in_sizes, int n_in,
                              void* d_out, int out_size, void* d_ws, size_t ws_size,
                              hipStream_t stream) {
    const int*   ids        = (const int*)d_in[0];
    const float* emb        = (const float*)d_in[1];
    const float* in_proj_w  = (const float*)d_in[2];
    const float* conv_w     = (const float*)d_in[3];
    const float* conv_b     = (const float*)d_in[4];
    const float* x_proj_w   = (const float*)d_in[5];
    const float* dt_proj_w  = (const float*)d_in[6];
    const float* dt_proj_b  = (const float*)d_in[7];
    const float* A_log      = (const float*)d_in[8];
    const float* Dp         = (const float*)d_in[9];
    const float* out_proj_w = (const float*)d_in[10];
    const float* norm_w     = (const float*)d_in[11];
    const float* norm_f_w   = (const float*)d_in[12];
    const float* lm_head_w  = (const float*)d_in[13];

    float* ws = (float*)d_ws;
    size_t o = 0;
    float* x    = ws + o; o += (size_t)SEQ_L * D_MODEL;                 // residual stream
    float* xn   = ws + o; o += (size_t)SEQ_L * D_MODEL;                 // normed
    float* xr   = ws + o; o += (size_t)SEQ_L * 2 * D_INNER;             // in_proj out (u|res)
    float* uc   = ws + o; o += (size_t)SEQ_L * D_INNER;                 // conv+silu
    float* mid  = ws + o; o += (size_t)SEQ_L * (DT_RANK + 2 * D_STATE); // x_proj out
    float* dtp  = ws + o; o += (size_t)SEQ_L * DT_PAD;                  // padded dt (A side)
    float* dtwp = ws + o; o += (size_t)N_LAYERS * DT_PAD * D_INNER;     // padded dt_proj_w
    float* dl   = ws + o; o += (size_t)SEQ_L * D_INNER;                 // delta
    float* y    = ws + o; o += (size_t)SEQ_L * D_INNER;                 // scan out

    const int EW = SEQ_L * D_INNER;

    // one-time padded copy of dt_proj weights (48 -> 64 rows, zeros below)
    pad_dtw_kernel<<<(N_LAYERS * DT_PAD * D_INNER + 255) / 256, 256, 0, stream>>>(
        dt_proj_w, dtwp);

    embed_kernel<<<SEQ_L, 256, 0, stream>>>(ids, emb, x);

    for (int i = 0; i < N_LAYERS; ++i) {
        rmsnorm_kernel<<<SEQ_L, 256, 0, stream>>>(x, norm_w + (size_t)i * D_MODEL, xn);

        // xr = xn @ W_in   (1024 x 3072, K=768)
        launch_gemm<false, 4>(xn, D_MODEL,
                              in_proj_w + (size_t)i * D_MODEL * 2 * D_INNER, 2 * D_INNER,
                              xr, 2 * D_INNER, nullptr, 0,
                              SEQ_L, 2 * D_INNER, D_MODEL, stream);

        conv_silu_kernel<<<(EW + 255) / 256, 256, 0, stream>>>(
            xr, conv_w + (size_t)i * D_CONV * D_INNER, conv_b + (size_t)i * D_INNER, uc);

        // mid = uc @ Wx   (1024 x 80, K=1536) -- N=80 uses the 1-tile variant
        launch_gemm<false, 1>(uc, D_INNER,
                              x_proj_w + (size_t)i * D_INNER * (DT_RANK + 2 * D_STATE),
                              DT_RANK + 2 * D_STATE,
                              mid, DT_RANK + 2 * D_STATE, nullptr, 0,
                              SEQ_L, DT_RANK + 2 * D_STATE, D_INNER, stream);

        // zero-pad dt columns, then delta_pre = dtp @ Wdt_padded (K=64)
        pad_dt_kernel<<<(SEQ_L * DT_PAD + 255) / 256, 256, 0, stream>>>(mid, dtp);
        launch_gemm<false, 4>(dtp, DT_PAD,
                              dtwp + (size_t)i * DT_PAD * D_INNER, D_INNER,
                              dl, D_INNER, nullptr, 0,
                              SEQ_L, D_INNER, DT_PAD, stream);

        softplus_kernel<<<(EW + 255) / 256, 256, 0, stream>>>(
            dl, dt_proj_b + (size_t)i * D_INNER);

        scan_kernel<<<(D_INNER + 255) / 256, 256, 0, stream>>>(
            dl, uc, mid, xr, A_log + (size_t)i * D_INNER * D_STATE,
            Dp + (size_t)i * D_INNER, y);

        // x = y @ W_out + x   (1024 x 768, K=1536, fused residual)
        launch_gemm<false, 4>(y, D_INNER,
                              out_proj_w + (size_t)i * D_INNER * D_MODEL, D_MODEL,
                              x, D_MODEL, x, D_MODEL,
                              SEQ_L, D_MODEL, D_INNER, stream);
    }

    // final norm -> lm_head -> logits = h1 @ embedding.T
    rmsnorm_kernel<<<SEQ_L, 256, 0, stream>>>(x, norm_f_w, xn);

    float* h1 = uc;  // reuse scratch
    launch_gemm<false, 4>(xn, D_MODEL, lm_head_w, D_MODEL,
                          h1, D_MODEL, nullptr, 0,
                          SEQ_L, D_MODEL, D_MODEL, stream);

    launch_gemm<true, 4>(h1, D_MODEL, emb, D_MODEL,
                         (float*)d_out, VOCAB, nullptr, 0,
                         SEQ_L, VOCAB, D_MODEL, stream);
}
